// HRGCN_12850542149723
// MI455X (gfx1250) — matrix operand order repaired and checked by reading, not compiled, not run on hardware
//
#include <hip/hip_runtime.h>
#include <hip/hip_bf16.h>

typedef __attribute__((ext_vector_type(16))) __bf16 v16bf;
typedef __attribute__((ext_vector_type(8)))  __bf16 v8bf;
typedef __attribute__((ext_vector_type(8)))  float  v8f;

__device__ __forceinline__ unsigned short f32_to_bf16_rne(float f) {
  union { float f; unsigned u; } v; v.f = f;
  unsigned u = v.u;
  u += 0x7FFFu + ((u >> 16) & 1u);   // round-to-nearest-even
  return (unsigned short)(u >> 16);
}

// ---------------------------------------------------------------- utilities
__global__ __launch_bounds__(256) void hr_zero_u32(unsigned* __restrict__ p, long n) {
  long i = (long)blockIdx.x * blockDim.x + threadIdx.x;
  if (i < n) p[i] = 0u;
}

__global__ __launch_bounds__(256) void hr_f32_to_bf16(const float* __restrict__ in,
                                                      unsigned short* __restrict__ out, long n) {
  long i = (long)blockIdx.x * blockDim.x + threadIdx.x;
  if (i < n) out[i] = f32_to_bf16_rne(in[i]);
}

__global__ __launch_bounds__(256) void hr_lrelu_to_bf16(const float* __restrict__ in,
                                                        unsigned short* __restrict__ out, long n) {
  long i = (long)blockIdx.x * blockDim.x + threadIdx.x;
  if (i < n) {
    float v = in[i];
    v = v > 0.0f ? v : 0.01f * v;
    out[i] = f32_to_bf16_rne(v);
  }
}

// counts over all 3 etypes at once; edges layout [ET, 2, E] (src row, dst row)
__global__ __launch_bounds__(256) void hr_count(const int* __restrict__ edges,
                                                int* __restrict__ cnt, int E, int N) {
  int i = blockIdx.x * blockDim.x + threadIdx.x;
  if (i >= 3 * E) return;
  int t = i / E;
  int k = i - t * E;
  int d = edges[(size_t)t * 2 * E + E + k];
  atomicAdd(&cnt[t * N + d], 1);
}

__global__ __launch_bounds__(256) void hr_invcnt(const int* __restrict__ cnt,
                                                 float* __restrict__ inv, int n) {
  int i = blockIdx.x * blockDim.x + threadIdx.x;
  if (i < n) {
    int c = cnt[i];
    inv[i] = c > 0 ? 1.0f / (float)c : 0.0f;
  }
}

// ---------------------------------------------------------------- WMMA GEMM
// C[M,H] = A[M,K](bf16) * Bw[H,K](bf16)^T + bias[H], one wave -> 16x64 tile.
__global__ __launch_bounds__(32) void hr_gemm_bf16_wmma(
    const __bf16* __restrict__ A,    // [M, K] row-major bf16
    const __bf16* __restrict__ Bw,   // [H, K] row-major bf16 (this etype's weight)
    const float*  __restrict__ bias, // [H]
    float*        __restrict__ C,    // [M, H]
    int K, int H) {
  const int lane = threadIdx.x & 31;
  const int r    = lane & 15;
  const int hf   = lane >> 4;
  const int m0   = blockIdx.x << 4;  // M tile base
  const int nb   = blockIdx.y << 6;  // N(=H) tile base, 64 wide

  // A fragment: lane holds row m0+r; half 0 -> K {k0..k0+15},
  //             half 1 -> K {k0+16..k0+31}
  const __bf16* arow = A + (size_t)(m0 + r) * K + 16 * hf;
  // B fragment: lane holds output column nb+j*16+r, 16 contiguous K per half
  const __bf16* bp0 = Bw + (size_t)(nb + r) * K + 16 * hf;
  const __bf16* bp1 = bp0 + (size_t)16 * K;
  const __bf16* bp2 = bp0 + (size_t)32 * K;
  const __bf16* bp3 = bp0 + (size_t)48 * K;

  v8f c0 = {}, c1 = {}, c2 = {}, c3 = {};

  for (int k0 = 0; k0 < K; k0 += 32) {
    v8bf alo = *(const v8bf*)(arow + k0);        // 16B
    v8bf ahi = *(const v8bf*)(arow + k0 + 8);    // 16B
    v16bf a;
#pragma unroll
    for (int j = 0; j < 8; ++j) { a[j] = alo[j]; a[j + 8] = ahi[j]; }
    v16bf b0 = *(const v16bf*)(bp0 + k0);        // 32B contiguous
    v16bf b1 = *(const v16bf*)(bp1 + k0);
    v16bf b2 = *(const v16bf*)(bp2 + k0);
    v16bf b3 = *(const v16bf*)(bp3 + k0);
    c0 = __builtin_amdgcn_wmma_f32_16x16x32_bf16(false, a, false, b0, (short)0, c0, false, false);
    c1 = __builtin_amdgcn_wmma_f32_16x16x32_bf16(false, a, false, b1, (short)0, c1, false, false);
    c2 = __builtin_amdgcn_wmma_f32_16x16x32_bf16(false, a, false, b2, (short)0, c2, false, false);
    c3 = __builtin_amdgcn_wmma_f32_16x16x32_bf16(false, a, false, b3, (short)0, c3, false, false);
  }

  const float bb0 = bias[nb + r];
  const float bb1 = bias[nb + 16 + r];
  const float bb2 = bias[nb + 32 + r];
  const float bb3 = bias[nb + 48 + r];
#pragma unroll
  for (int i = 0; i < 8; ++i) {
    const size_t m = (size_t)(m0 + (hf << 3) + i);   // VGPR i: M=i (lanes 0-15), M=8+i (16-31)
    float* crow = C + m * H + nb + r;
    crow[0]  = c0[i] + bb0;
    crow[16] = c1[i] + bb1;
    crow[32] = c2[i] + bb2;
    crow[48] = c3[i] + bb3;
  }
}

// ---------------------------------------------------------------- scatter-mean
// out[dst] += Wh[src] * inv_indeg[dst]; one thread = one edge x 4 features.
__global__ __launch_bounds__(256) void hr_scatter_mean(
    const float* __restrict__ Wh, const int* __restrict__ src, const int* __restrict__ dst,
    const float* __restrict__ inv, float* __restrict__ out, int E, int D) {
  int tid = blockIdx.x * blockDim.x + threadIdx.x;
  const int per = D >> 2;
  if (tid >= E * per) return;
  int ei = tid / per;
  int c  = tid - ei * per;
  int s = src[ei];
  int d = dst[ei];
  const float4 v = ((const float4*)(Wh + (size_t)s * D))[c];
  const float  w = inv[d];
  float* o = out + (size_t)d * D + (c << 2);
  unsafeAtomicAdd(o + 0, v.x * w);
  unsafeAtomicAdd(o + 1, v.y * w);
  unsafeAtomicAdd(o + 2, v.z * w);
  unsafeAtomicAdd(o + 3, v.w * w);
}

// ---------------------------------------------------------------- launch
extern "C" void kernel_launch(void* const* d_in, const int* in_sizes, int n_in,
                              void* d_out, int out_size, void* d_ws, size_t ws_size,
                              hipStream_t stream) {
  (void)in_sizes; (void)n_in; (void)out_size; (void)ws_size;
  const int N = 50000, E = 300000, ET = 3, DIN = 256, DHID = 256, DOUT = 128;

  const float* x  = (const float*)d_in[0];
  const int*   e1 = (const int*)d_in[1];
  const int*   e2 = (const int*)d_in[2];
  const float* W1 = (const float*)d_in[3];
  const float* b1 = (const float*)d_in[4];
  const float* W2 = (const float*)d_in[5];
  const float* b2 = (const float*)d_in[6];
  float* out = (float*)d_out;

  char* ws = (char*)d_ws;
  size_t off = 0;
  auto alloc = [&](size_t bytes) -> void* {
    void* p = ws + off;
    off += (bytes + 255) & ~(size_t)255;
    return p;
  };
  unsigned short* xbf  = (unsigned short*)alloc((size_t)N * DIN * 2);
  unsigned short* hbf  = (unsigned short*)alloc((size_t)N * DHID * 2);
  float*          Wh   = (float*)alloc((size_t)N * DHID * 4);      // reused [N,128] in layer 2
  float*          hacc = (float*)alloc((size_t)N * DHID * 4);
  int*            cnt  = (int*)alloc((size_t)ET * N * 4);
  float*          inv  = (float*)alloc((size_t)ET * N * 4);
  unsigned short* w1bf = (unsigned short*)alloc((size_t)ET * DHID * DIN * 2);
  unsigned short* w2bf = (unsigned short*)alloc((size_t)ET * DOUT * DHID * 2);

  auto cdiv = [](long a, long b) { return (int)((a + b - 1) / b); };

  // init accumulators / counts
  hr_zero_u32<<<cdiv((long)N * DHID, 256), 256, 0, stream>>>((unsigned*)hacc, (long)N * DHID);
  hr_zero_u32<<<cdiv((long)N * DOUT, 256), 256, 0, stream>>>((unsigned*)out, (long)N * DOUT);
  hr_zero_u32<<<cdiv((long)ET * N, 256), 256, 0, stream>>>((unsigned*)cnt, (long)ET * N);

  // bf16 conversions
  hr_f32_to_bf16<<<cdiv((long)N * DIN, 256), 256, 0, stream>>>(x, xbf, (long)N * DIN);
  hr_f32_to_bf16<<<cdiv((long)ET * DHID * DIN, 256), 256, 0, stream>>>(W1, w1bf, (long)ET * DHID * DIN);
  hr_f32_to_bf16<<<cdiv((long)ET * DOUT * DHID, 256), 256, 0, stream>>>(W2, w2bf, (long)ET * DOUT * DHID);

  // ---- layer 1
  hr_count<<<cdiv((long)ET * E, 256), 256, 0, stream>>>(e1, cnt, E, N);
  hr_invcnt<<<cdiv((long)ET * N, 256), 256, 0, stream>>>(cnt, inv, ET * N);
  for (int t = 0; t < ET; ++t) {
    dim3 g(N / 16, DHID / 64);
    hr_gemm_bf16_wmma<<<g, 32, 0, stream>>>(
        (const __bf16*)xbf, (const __bf16*)(w1bf + (size_t)t * DHID * DIN),
        b1 + (size_t)t * DHID, Wh, DIN, DHID);
    const int* src = e1 + (size_t)t * 2 * E;
    const int* dst = src + E;
    hr_scatter_mean<<<cdiv((long)E * (DHID / 4), 256), 256, 0, stream>>>(
        Wh, src, dst, inv + (size_t)t * N, hacc, E, DHID);
  }
  hr_lrelu_to_bf16<<<cdiv((long)N * DHID, 256), 256, 0, stream>>>(hacc, hbf, (long)N * DHID);

  // ---- layer 2
  hr_zero_u32<<<cdiv((long)ET * N, 256), 256, 0, stream>>>((unsigned*)cnt, (long)ET * N);
  hr_count<<<cdiv((long)ET * E, 256), 256, 0, stream>>>(e2, cnt, E, N);
  hr_invcnt<<<cdiv((long)ET * N, 256), 256, 0, stream>>>(cnt, inv, ET * N);
  for (int t = 0; t < ET; ++t) {
    dim3 g(N / 16, DOUT / 64);
    hr_gemm_bf16_wmma<<<g, 32, 0, stream>>>(
        (const __bf16*)hbf, (const __bf16*)(w2bf + (size_t)t * DOUT * DHID),
        b2 + (size_t)t * DOUT, Wh, DHID, DOUT);
    const int* src = e2 + (size_t)t * 2 * E;
    const int* dst = src + E;
    hr_scatter_mean<<<cdiv((long)E * (DOUT / 4), 256), 256, 0, stream>>>(
        Wh, src, dst, inv + (size_t)t * N, out, E, DOUT);
  }
}
